// SAModule_30253749633671
// MI455X (gfx1250) — matrix operand order repaired and checked by reading, not compiled, not run on hardware
//
#include <hip/hip_runtime.h>

// ---- static problem config (mirrors the reference) ----
#define NB    8        // clouds
#define NP    2048     // points per cloud
#define NM    1024     // FPS samples per cloud
#define NK    128      // neighbors per center
#define FIN   32       // input feature channels
#define NH    64       // hidden/out channels
#define NC    (NB*NM)  // total centers = 8192
#define RR2   0.04f    // radius^2

typedef __attribute__((ext_vector_type(16))) _Float16 v16h;
typedef __attribute__((ext_vector_type(8)))  _Float16 v8h;
typedef __attribute__((ext_vector_type(8)))  float    v8f;

__device__ __forceinline__ v8h zero8() {
  v8h z;
#pragma unroll
  for (int i = 0; i < 8; ++i) z[i] = (_Float16)0.0f;
  return z;
}

__device__ __forceinline__ v8f splat8(float v) {
  v8f r;
#pragma unroll
  for (int i = 0; i < 8; ++i) r[i] = v;
  return r;
}

// Build a 16-half WMMA fragment from two 16-byte LDS reads (ds_load_b128 x2).
__device__ __forceinline__ v16h ld_frag(const _Float16* lo, const _Float16* hi) {
  v8h a = *(const v8h*)lo;
  v8h b = *(const v8h*)hi;
  v16h r;
#pragma unroll
  for (int i = 0; i < 8; ++i) { r[i] = a[i]; r[8 + i] = b[i]; }
  return r;
}

// ---------------------------------------------------------------------------
// Kernel 1: farthest point sampling, one block per cloud.
// Matches reference: sel[0]=0; nxt = argmax(d) (first occurrence); d=min(d,d_nxt)
// ---------------------------------------------------------------------------
__global__ void __launch_bounds__(256) fps_kernel(const float* __restrict__ pos,
                                                  int* __restrict__ sel) {
  const int b   = blockIdx.x;
  const int tid = threadIdx.x;
  const int lane = tid & 31, wid = tid >> 5;
  __shared__ float s_v[8];
  __shared__ int   s_i[8];
  __shared__ int   s_best;

  const float* P = pos + (size_t)b * NP * 3;
  float px[8], py[8], pz[8], d[8];
  const float x0 = P[0], y0 = P[1], z0 = P[2];
#pragma unroll
  for (int k = 0; k < 8; ++k) {
    const int jj = tid + k * 256;
    px[k] = P[jj * 3 + 0]; py[k] = P[jj * 3 + 1]; pz[k] = P[jj * 3 + 2];
    const float dx = px[k] - x0, dy = py[k] - y0, dz = pz[k] - z0;
    d[k] = dx * dx + dy * dy + dz * dz;
  }
  if (tid == 0) sel[b * NM + 0] = 0;

  for (int i = 1; i < NM; ++i) {
    float bv = d[0];
    int   bi = tid;
#pragma unroll
    for (int k = 1; k < 8; ++k) {
      const int jj = tid + k * 256;
      if (d[k] > bv || (d[k] == bv && jj < bi)) { bv = d[k]; bi = jj; }
    }
    // wave32 argmax (tie -> smaller index, like jnp.argmax)
    for (int off = 16; off > 0; off >>= 1) {
      const float ov = __shfl_xor(bv, off, 32);
      const int   oi = __shfl_xor(bi, off, 32);
      if (ov > bv || (ov == bv && oi < bi)) { bv = ov; bi = oi; }
    }
    if (lane == 0) { s_v[wid] = bv; s_i[wid] = bi; }
    __syncthreads();
    if (tid == 0) {
      float vv = s_v[0]; int ii = s_i[0];
      for (int q = 1; q < 8; ++q)
        if (s_v[q] > vv || (s_v[q] == vv && s_i[q] < ii)) { vv = s_v[q]; ii = s_i[q]; }
      s_best = ii;
      sel[b * NM + i] = ii;
    }
    __syncthreads();
    const int bidx = s_best;
    const float bx = P[bidx * 3 + 0], by = P[bidx * 3 + 1], bz = P[bidx * 3 + 2];
#pragma unroll
    for (int k = 0; k < 8; ++k) {
      const float dx = px[k] - bx, dy = py[k] - by, dz = pz[k] - bz;
      d[k] = fminf(d[k], dx * dx + dy * dy + dz * dz);
    }
  }
}

// ---------------------------------------------------------------------------
// Kernel 2: ball query — one wave32 per center, first K valid by index
// (ballot + prefix popcount preserves index order exactly).
// Also emits pos_out / batch_out / y_idx / x_idx (ints as float values).
// ---------------------------------------------------------------------------
__global__ void __launch_bounds__(256) ball_kernel(const float* __restrict__ pos,
                                                   const int* __restrict__ sel,
                                                   int* __restrict__ nbr,
                                                   int* __restrict__ cntArr,
                                                   float* __restrict__ outPos,
                                                   float* __restrict__ outBatch,
                                                   float* __restrict__ outXidx,
                                                   float* __restrict__ outYidx) {
  const int w = threadIdx.x >> 5, lane = threadIdx.x & 31;
  const int c = blockIdx.x * 8 + w;          // 0..8191
  const int b = c >> 10, m = c & 1023;
  const float* P = pos + (size_t)b * NP * 3;
  const int si = sel[b * NM + m];
  const float cx = P[si * 3 + 0], cy = P[si * 3 + 1], cz = P[si * 3 + 2];
  if (lane == 0) {
    outPos[c * 3 + 0] = cx; outPos[c * 3 + 1] = cy; outPos[c * 3 + 2] = cz;
    outBatch[c] = (float)b;
    outYidx[c]  = (float)c;
  }
  int cnt = 0;
  for (int j0 = 0; j0 < NP; j0 += 32) {
    const int j = j0 + lane;
    const float dx = P[j * 3 + 0] - cx;
    const float dy = P[j * 3 + 1] - cy;
    const float dz = P[j * 3 + 2] - cz;
    const bool valid = (dx * dx + dy * dy + dz * dz) <= RR2;
    const unsigned mask = (unsigned)__ballot(valid);
    const int p = cnt + __popc(mask & ((1u << lane) - 1u));
    if (valid && p < NK) {
      nbr[(size_t)c * NK + p]     = j;
      outXidx[(size_t)c * NK + p] = (float)(b * NP + j);
    }
    cnt += __popc(mask);
    if (cnt >= NK) break;                    // uniform across the wave
  }
  const int cc = min(cnt, NK);
  if (lane == 0) cntArr[c] = cc;
  for (int p = cc + lane; p < NK; p += 32) {
    nbr[(size_t)c * NK + p]     = 0;         // clamped index for gather
    outXidx[(size_t)c * NK + p] = -1.0f;     // padding marker
  }
}

// ---------------------------------------------------------------------------
// Kernel 3: PointConv MLP via v_wmma_f32_16x16x32_f16, one wave per center.
// Per 16-edge tile: layer1 = 2 K-chunks x 4 N-tiles, layer2 = same, then
// ReLU + row-masked max aggregation. A/B fragments follow ISA 7.12.2 layouts.
// ---------------------------------------------------------------------------
__global__ void __launch_bounds__(256) conv_kernel(const float* __restrict__ X,
                                                   const float* __restrict__ POS,
                                                   const float* __restrict__ W1,
                                                   const float* __restrict__ B1,
                                                   const float* __restrict__ W2,
                                                   const float* __restrict__ B2,
                                                   const int* __restrict__ nbr,
                                                   const int* __restrict__ cntArr,
                                                   const float* __restrict__ outPos,
                                                   float* __restrict__ Xout) {
  // Weights stored K-major ("transposed"): sWt[col*64 + k]; W1 K padded 35->64.
  __shared__ __attribute__((aligned(16))) _Float16 sWt1[64 * 64];
  __shared__ __attribute__((aligned(16))) _Float16 sWt2[64 * 64];
  __shared__ float sB1[64], sB2[64];
  __shared__ __attribute__((aligned(16))) _Float16 sFeat [8][16 * 64]; // per-wave A tile, layer1
  __shared__ __attribute__((aligned(16))) _Float16 sFeat2[8][16 * 64]; // per-wave A tile, layer2

  const int tid = threadIdx.x;
  for (int idx = tid; idx < 64 * 64; idx += 256) {
    const int col = idx >> 6, k = idx & 63;
    sWt1[idx] = (_Float16)((k < (FIN + 3)) ? W1[k * NH + col] : 0.0f);
    sWt2[idx] = (_Float16)W2[k * NH + col];
  }
  if (tid < 64) { sB1[tid] = B1[tid]; sB2[tid] = B2[tid]; }
  __syncthreads();

  const int w = tid >> 5, lane = tid & 31;
  const int c = blockIdx.x * 8 + w;          // one center per wave
  const int bIdx = c >> 10;
  const int r = lane & 15, part = lane >> 4; // A-row / K-half split (ISA layout)

  const float cx = outPos[c * 3 + 0];
  const float cy = outPos[c * 3 + 1];
  const float cz = outPos[c * 3 + 2];
  const int cc = cntArr[c];

  _Float16* ft1 = sFeat[w];
  _Float16* ft2 = sFeat2[w];
  float rmax[4] = {-1e30f, -1e30f, -1e30f, -1e30f};

  for (int t = 0; t < 8; ++t) {
    // ---- build 16x64 f16 A-tile: [x_j(32) | rel(3) | zeros(29)] ----
    const int j = nbr[(size_t)c * NK + t * 16 + r];
    _Float16* frow = ft1 + r * 64;
    const v8h z = zero8();
    *(v8h*)(frow + 32 + part * 16)     = z;   // zero pad K=32..63
    *(v8h*)(frow + 32 + part * 16 + 8) = z;
    const float* xr = X + ((size_t)(bIdx * NP + j)) * FIN + part * 16;
    const float4 f0 = ((const float4*)xr)[0];
    const float4 f1 = ((const float4*)xr)[1];
    const float4 f2 = ((const float4*)xr)[2];
    const float4 f3 = ((const float4*)xr)[3];
    const v8h h0 = { (_Float16)f0.x, (_Float16)f0.y, (_Float16)f0.z, (_Float16)f0.w,
                     (_Float16)f1.x, (_Float16)f1.y, (_Float16)f1.z, (_Float16)f1.w };
    const v8h h1 = { (_Float16)f2.x, (_Float16)f2.y, (_Float16)f2.z, (_Float16)f2.w,
                     (_Float16)f3.x, (_Float16)f3.y, (_Float16)f3.z, (_Float16)f3.w };
    *(v8h*)(frow + part * 16)     = h0;       // x features K = part*16 .. +16
    *(v8h*)(frow + part * 16 + 8) = h1;
    if (part) {                               // rel pos at K=32..34 (after zeros, in-order)
      const float* pr = POS + ((size_t)(bIdx * NP + j)) * 3;
      frow[32] = (_Float16)(pr[0] - cx);
      frow[33] = (_Float16)(pr[1] - cy);
      frow[34] = (_Float16)(pr[2] - cz);
    }

    // ---- layer 1: h = relu(A(16x64) * W1(64x64) + b1) ----
    v8f acc[4];
#pragma unroll
    for (int nt = 0; nt < 4; ++nt) acc[nt] = splat8(sB1[nt * 16 + r]);
#pragma unroll
    for (int ch = 0; ch < 2; ++ch) {
      const v16h a = ld_frag(frow + ch * 32 + part * 8,
                             frow + ch * 32 + 16 + part * 8);
#pragma unroll
      for (int nt = 0; nt < 4; ++nt) {
        const _Float16* wc = sWt1 + (nt * 16 + r) * 64 + ch * 32 + part * 8;
        const v16h bf = ld_frag(wc, wc + 16);
        acc[nt] = __builtin_amdgcn_wmma_f32_16x16x32_f16(
            false, a, false, bf, (short)0, acc[nt], false, false);
      }
    }
    // ReLU + stage h as layer-2 A-tile (D layout: col = nt*16+r, row = part*8+v)
#pragma unroll
    for (int nt = 0; nt < 4; ++nt) {
#pragma unroll
      for (int v = 0; v < 8; ++v) {
        const float hv = fmaxf(acc[nt][v], 0.0f);
        ft2[(part * 8 + v) * 64 + (nt * 16 + r)] = (_Float16)hv;
      }
    }

    // ---- layer 2: h2 = relu(H(16x64) * W2(64x64) + b2) ----
    _Float16* frow2 = ft2 + r * 64;
    v8f acc2[4];
#pragma unroll
    for (int nt = 0; nt < 4; ++nt) acc2[nt] = splat8(sB2[nt * 16 + r]);
#pragma unroll
    for (int ch = 0; ch < 2; ++ch) {
      const v16h a = ld_frag(frow2 + ch * 32 + part * 8,
                             frow2 + ch * 32 + 16 + part * 8);
#pragma unroll
      for (int nt = 0; nt < 4; ++nt) {
        const _Float16* wc = sWt2 + (nt * 16 + r) * 64 + ch * 32 + part * 8;
        const v16h bf = ld_frag(wc, wc + 16);
        acc2[nt] = __builtin_amdgcn_wmma_f32_16x16x32_f16(
            false, a, false, bf, (short)0, acc2[nt], false, false);
      }
    }
    // ReLU + row-masked running max (mask: edge slot < cc)
#pragma unroll
    for (int nt = 0; nt < 4; ++nt) {
#pragma unroll
      for (int v = 0; v < 8; ++v) {
        const int grow = t * 16 + part * 8 + v;
        const float hv = fmaxf(acc2[nt][v], 0.0f);
        if (grow < cc) rmax[nt] = fmaxf(rmax[nt], hv);
      }
    }
  }

  // merge the two row-halves (lanes l <-> l+16) and store 64 outputs/center
#pragma unroll
  for (int nt = 0; nt < 4; ++nt) {
    const float other = __shfl_xor(rmax[nt], 16, 32);
    float full = fmaxf(rmax[nt], other);
    full = fmaxf(full, 0.0f);                // -inf (no valid rows) -> 0, relu'd vals unchanged
    if (part == 0) Xout[(size_t)c * NH + nt * 16 + r] = full;
  }
}

// ---------------------------------------------------------------------------
extern "C" void kernel_launch(void* const* d_in, const int* in_sizes, int n_in,
                              void* d_out, int out_size, void* d_ws, size_t ws_size,
                              hipStream_t stream) {
  (void)in_sizes; (void)n_in; (void)out_size; (void)ws_size;
  const float* x   = (const float*)d_in[0];
  const float* pos = (const float*)d_in[1];
  // d_in[2] = batch (implied by layout, unused)
  const float* W1  = (const float*)d_in[3];
  const float* b1  = (const float*)d_in[4];
  const float* W2  = (const float*)d_in[5];
  const float* b2  = (const float*)d_in[6];

  float* out      = (float*)d_out;
  float* xOut     = out;                         // [8192, 64]
  float* posOut   = out + (size_t)NC * NH;       // [8192, 3]
  float* batchOut = posOut + (size_t)NC * 3;     // [8192]
  float* xidxOut  = batchOut + NC;               // [8192, 128]
  float* yidxOut  = xidxOut + (size_t)NC * NK;   // [8192]

  char* ws  = (char*)d_ws;
  int* sel  = (int*)ws;                          // [8192]
  int* cnt  = (int*)(ws + (size_t)NC * 4);       // [8192]
  int* nbrI = (int*)(ws + (size_t)2 * NC * 4);   // [8192, 128]

  fps_kernel <<<NB, 256, 0, stream>>>(pos, sel);
  ball_kernel<<<NC / 8, 256, 0, stream>>>(pos, sel, nbrI, cnt,
                                          posOut, batchOut, xidxOut, yidxOut);
  conv_kernel<<<NC / 8, 256, 0, stream>>>(x, pos, W1, b1, W2, b2,
                                          nbrI, cnt, posOut, xOut);
}
